// vrnn_38946763440808
// MI455X (gfx1250) — compile-verified, matrix operand order checked by reading, not run on hardware
//
#include <hip/hip_runtime.h>
#include <hip/hip_bf16.h>
#include <stdint.h>

// ---------------------------------------------------------------- constants
#define SEQL   192
#define BATCH  256
#define HIDN   512
#define XD     128
#define LOUT   24
#define EPSF           1.1920929e-7f
#define HALF_LOG_2PI_F 0.9189385332046727f

// ---------------------------------------------------------------- types
typedef __attribute__((ext_vector_type(16))) __bf16        v16bf;
typedef __attribute__((ext_vector_type(8)))  float         v8f;
typedef __attribute__((ext_vector_type(4)))  unsigned int  u32x4;
typedef int v4i_ls __attribute__((vector_size(16)));       // builtin's pointee type

union Frag16 { v16bf v; u32x4 q[2]; };

__device__ __forceinline__ unsigned short f2bf(float f) {
    unsigned int u = __float_as_uint(f);
    u = (u + 0x7FFFu + ((u >> 16) & 1u)) >> 16;   // round-to-nearest-even
    return (unsigned short)u;
}
__device__ __forceinline__ float softplusf(float x) {
    return fmaxf(x, 0.0f) + log1pf(expf(-fabsf(x)));
}
__device__ __forceinline__ float sigmf(float x) {
    return 1.0f / (1.0f + expf(-x));
}

// gfx1250 async global->LDS path (ASYNCcnt), guarded so compile never breaks
#if __has_builtin(__builtin_amdgcn_global_load_async_to_lds_b128) && \
    __has_builtin(__builtin_amdgcn_s_wait_asynccnt)
#define GEMM_ASYNC 1
__device__ __forceinline__ void async_b128(const unsigned short* g, unsigned short* l) {
    __builtin_amdgcn_global_load_async_to_lds_b128(
        (__attribute__((address_space(1))) v4i_ls*)g,
        (__attribute__((address_space(3))) v4i_ls*)l, 0, 0);
}
#else
#define GEMM_ASYNC 0
#endif

// ---------------------------------------------------------------- WMMA GEMM
// C[M,N] = act( A[M,K](bf16) * W[N,K]^T(bf16) + bias ),  A split at K0
// (A = [A0 | A1] along K) so concat(x,h) @ W needs no materialized concat.
// Block: 256 threads (8 wave32), 64x64 output tile, BK=64, double-buffered
// LDS; each wave owns a 32x16 strip -> 4 v_wmma_f32_16x16x32_bf16 per stage.
#define LSTR 72   // LDS row stride (bf16): 144B rows -> conflict-free b128 frags
#define BKT  64

template<bool RELU, bool OUTB, bool OUTF>
__global__ __launch_bounds__(256) void gemm_wmma_bf16(
    const unsigned short* __restrict__ A0, const unsigned short* __restrict__ A1,
    int K0, int K, const unsigned short* __restrict__ W,
    const float* __restrict__ bias,
    unsigned short* __restrict__ outb, float* __restrict__ outf, int N)
{
    __shared__ __align__(16) unsigned short As[2][64 * LSTR];
    __shared__ __align__(16) unsigned short Bs[2][64 * LSTR];

    const int tid  = threadIdx.x;
    const int lane = tid & 31;
    const int wv   = tid >> 5;           // wave id 0..7 (wave32)
    const int wm   = (wv & 1) * 32;      // wave M offset in block
    const int wn   = (wv >> 1) * 16;     // wave N offset in block
    const int bm   = blockIdx.y * 64;
    const int bn   = blockIdx.x * 64;

    const int row0 = tid >> 3;           // cooperative tile load: rows r, r+32
    const int colg = (tid & 7) * 8;      // 8 x 8 bf16 (16B) per row
    const int lm   = lane & 15;
    const int lk   = (lane >> 4) * 8;

    v8f acc0 = {}; v8f acc1 = {};

    auto gptrA = [&](int kt, int r) -> const unsigned short* {
        const unsigned short* Ap; int ldA; int kk;
        if (kt < K0) { Ap = A0; ldA = K0;     kk = kt;      }
        else         { Ap = A1; ldA = K - K0; kk = kt - K0; }
        return Ap + (size_t)(bm + r) * ldA + kk + colg;
    };
    auto gptrW = [&](int kt, int r) -> const unsigned short* {
        return W + (size_t)(bn + r) * K + kt + colg;
    };

#if GEMM_ASYNC
    auto issue = [&](int kt, int buf) {
        async_b128(gptrA(kt, row0),      &As[buf][row0 * LSTR + colg]);
        async_b128(gptrA(kt, row0 + 32), &As[buf][(row0 + 32) * LSTR + colg]);
        async_b128(gptrW(kt, row0),      &Bs[buf][row0 * LSTR + colg]);
        async_b128(gptrW(kt, row0 + 32), &Bs[buf][(row0 + 32) * LSTR + colg]);
    };
    issue(0, 0);
    __builtin_amdgcn_s_wait_asynccnt(0);
    __syncthreads();
#else
    u32x4 ra0, ra1, rw0, rw1;
    auto issue = [&](int kt) {
        ra0 = *(const u32x4*)gptrA(kt, row0);
        ra1 = *(const u32x4*)gptrA(kt, row0 + 32);
        rw0 = *(const u32x4*)gptrW(kt, row0);
        rw1 = *(const u32x4*)gptrW(kt, row0 + 32);
    };
    auto commit = [&](int buf) {
        *(u32x4*)&As[buf][row0 * LSTR + colg]        = ra0;
        *(u32x4*)&As[buf][(row0 + 32) * LSTR + colg] = ra1;
        *(u32x4*)&Bs[buf][row0 * LSTR + colg]        = rw0;
        *(u32x4*)&Bs[buf][(row0 + 32) * LSTR + colg] = rw1;
    };
    issue(0); commit(0);
    __syncthreads();
#endif

    const int nTiles = K >> 6;
    for (int it = 0; it < nTiles; ++it) {
        const int cur = it & 1;
#if GEMM_ASYNC
        if (it + 1 < nTiles) issue((it + 1) << 6, cur ^ 1);   // overlap with WMMA
#else
        if (it + 1 < nTiles) issue((it + 1) << 6);            // global -> regs early
#endif
        const unsigned short* apb = &As[cur][(wm + lm) * LSTR + lk];
        const unsigned short* bpb = &Bs[cur][(wn + lm) * LSTR + lk];
#pragma unroll
        for (int ks = 0; ks < BKT; ks += 32) {
            Frag16 fb, fa0, fa1;
            fb.q[0]  = *(const u32x4*)(bpb + ks);
            fb.q[1]  = *(const u32x4*)(bpb + ks + 16);
            fa0.q[0] = *(const u32x4*)(apb + ks);
            fa0.q[1] = *(const u32x4*)(apb + ks + 16);
            fa1.q[0] = *(const u32x4*)(apb + 16 * LSTR + ks);
            fa1.q[1] = *(const u32x4*)(apb + 16 * LSTR + ks + 16);
            acc0 = __builtin_amdgcn_wmma_f32_16x16x32_bf16(false, fa0.v, false, fb.v,
                                                           (short)0, acc0, false, false);
            acc1 = __builtin_amdgcn_wmma_f32_16x16x32_bf16(false, fa1.v, false, fb.v,
                                                           (short)0, acc1, false, false);
        }
#if GEMM_ASYNC
        __builtin_amdgcn_s_wait_asynccnt(0);
        __syncthreads();
#else
        if (it + 1 < nTiles) commit(cur ^ 1);                 // waits loadcnt here
        __syncthreads();
#endif
    }

    // epilogue: C/D layout -> lane<16: m=r, n=lane ; lane>=16: m=8+r, n=lane-16
    const int   n  = bn + wn + lm;
    const float bv = bias ? bias[n] : 0.0f;
    const int   mb = bm + wm + (lane >> 4) * 8;
#pragma unroll
    for (int r = 0; r < 8; ++r) {
        float v0 = acc0[r] + bv;
        float v1 = acc1[r] + bv;
        if (RELU) { v0 = fmaxf(v0, 0.0f); v1 = fmaxf(v1, 0.0f); }
        size_t i0 = (size_t)(mb + r)      * N + n;
        size_t i1 = (size_t)(mb + r + 16) * N + n;
        if (OUTB) { outb[i0] = f2bf(v0); outb[i1] = f2bf(v1); }
        if (OUTF) { outf[i0] = v0;       outf[i1] = v1;       }
    }
}

// ---------------------------------------------------------------- small kernels
__global__ void pack_bf16_k(const float* __restrict__ s, unsigned short* __restrict__ d, int n) {
    int i = blockIdx.x * 256 + threadIdx.x;
    if (i < n) d[i] = f2bf(s[i]);
}
__global__ void copy_f32_k(const float* __restrict__ s, float* __restrict__ d, int n) {
    int i = blockIdx.x * 256 + threadIdx.x;
    if (i < n) d[i] = s[i];
}
__global__ void init_state_k(float* h0f, float* h1f, unsigned short* h0b,
                             unsigned short* h1b, float* acc) {
    int i = blockIdx.x * 256 + threadIdx.x;
    if (i < BATCH * HIDN) { h0f[i] = 0.f; h1f[i] = 0.f; h0b[i] = 0; h1b[i] = 0; }
    if (i < 8) acc[i] = 0.f;
}
// x = concat(X_lag, X_cov, emb0[idx0], emb1[idx1]) -> bf16 [S*B,128]
__global__ void build_x_k(const int* __restrict__ Xe, const float* __restrict__ Xc,
                          const float* __restrict__ Xl, const float* __restrict__ e0,
                          const float* __restrict__ e1, unsigned short* __restrict__ xb) {
    int i  = blockIdx.x * 256 + threadIdx.x;   // over S*B*128, exact multiple
    int j  = i & 127;
    int sb = i >> 7;
    float v;
    if      (j < 64)  v = Xl[(size_t)sb * 64 + j];
    else if (j < 104) v = Xc[(size_t)sb * 40 + (j - 64)];
    else if (j < 120) v = e0[(size_t)Xe[(size_t)sb * 2 + 0] * 16 + (j - 104)];
    else              v = e1[(size_t)Xe[(size_t)sb * 2 + 1] * 8  + (j - 120)];
    xb[i] = f2bf(v);
}
// z-sample + KLD partial sum.  encms/priorms: [B,1024] = [mean | std_lin]
__global__ __launch_bounds__(256) void zkld_k(const float* __restrict__ encms,
                                              const float* __restrict__ priorms,
                                              const float* __restrict__ eps_t,
                                              unsigned short* __restrict__ zb,
                                              float* __restrict__ acc) {
    int i = blockIdx.x * 256 + threadIdx.x;    // over B*512, exact
    int b = i >> 9, j = i & 511;
    float em = encms  [(size_t)b * 1024 + j];
    float es = softplusf(encms  [(size_t)b * 1024 + 512 + j]);
    float pm = priorms[(size_t)b * 1024 + j];
    float ps = softplusf(priorms[(size_t)b * 1024 + 512 + j]);
    float z  = eps_t[i] * es + em;
    zb[i] = f2bf(z);
    float d  = em - pm;
    float kl = 0.5f * (2.f * logf(ps + EPSF) - 2.f * logf(es + EPSF)
                       + (es * es + d * d) / (ps * ps) - 1.f);
    __shared__ float red[256];
    red[threadIdx.x] = kl; __syncthreads();
    for (int s = 128; s > 0; s >>= 1) {
        if (threadIdx.x < s) red[threadIdx.x] += red[threadIdx.x + s];
        __syncthreads();
    }
    if (threadIdx.x == 0) atomicAdd(&acc[0], red[0]);
}
// NLL partial sums + streaming means/stds to d_out for the last LOUT steps
__global__ __launch_bounds__(256) void nll_k(const float* __restrict__ decms,
                                             const int* __restrict__ Xe,
                                             const float* __restrict__ Xc,
                                             const float* __restrict__ Xl,
                                             const float* __restrict__ e0,
                                             const float* __restrict__ e1,
                                             int t, float* __restrict__ acc,
                                             float* __restrict__ dout) {
    int i = blockIdx.x * 256 + threadIdx.x;    // over B*128, exact
    int b = i >> 7, j = i & 127;
    float mean = decms[(size_t)b * 256 + j];
    float sd   = softplusf(decms[(size_t)b * 256 + 128 + j]);
    int sb = t * BATCH + b;
    float x;
    if      (j < 64)  x = Xl[(size_t)sb * 64 + j];
    else if (j < 104) x = Xc[(size_t)sb * 40 + (j - 64)];
    else if (j < 120) x = e0[(size_t)Xe[(size_t)sb * 2 + 0] * 16 + (j - 104)];
    else              x = e1[(size_t)Xe[(size_t)sb * 2 + 1] * 8  + (j - 120)];
    float d  = x - mean;
    float nl = logf(sd + EPSF) + HALF_LOG_2PI_F + d * d / (2.f * sd * sd);
    if (t >= SEQL - LOUT) {
        size_t o = 2 + (size_t)(t - (SEQL - LOUT)) * BATCH * XD + i;
        dout[o] = mean;
        dout[o + (size_t)LOUT * BATCH * XD] = sd;
    }
    __shared__ float red[256];
    red[threadIdx.x] = nl; __syncthreads();
    for (int s = 128; s > 0; s >>= 1) {
        if (threadIdx.x < s) red[threadIdx.x] += red[threadIdx.x + s];
        __syncthreads();
    }
    if (threadIdx.x == 0) {
        atomicAdd(&acc[1], red[0]);
        if (t >= SEQL - LOUT) atomicAdd(&acc[2], red[0]);
    }
}
// GRU gate combine: gi,gh = [B,1536] (r|z|n chunks of 512), updates h (f32+bf16)
__global__ __launch_bounds__(256) void gru_comb_k(const float* __restrict__ gi,
                                                  const float* __restrict__ gh,
                                                  float* __restrict__ hf,
                                                  unsigned short* __restrict__ hb) {
    int i = blockIdx.x * 256 + threadIdx.x;    // over B*512, exact
    int b = i >> 9, j = i & 511;
    const float* gib = gi + (size_t)b * 1536;
    const float* ghb = gh + (size_t)b * 1536;
    float r  = sigmf(gib[j]        + ghb[j]);
    float z  = sigmf(gib[512 + j]  + ghb[512 + j]);
    float nn = tanhf(gib[1024 + j] + r * ghb[1024 + j]);
    float hn = (1.f - z) * nn + z * hf[i];
    hf[i] = hn;
    hb[i] = f2bf(hn);
}
__global__ void fin_k(const float* __restrict__ acc, float* __restrict__ dout) {
    if (threadIdx.x == 0) {
        dout[0] = acc[0] + acc[1];   // total = KLD + NLL
        dout[1] = acc[2];            // pred_y_loss = NLL over last LOUT steps
    }
}

// ---------------------------------------------------------------- launch
extern "C" void kernel_launch(void* const* d_in, const int* in_sizes, int n_in,
                              void* d_out, int out_size, void* d_ws, size_t ws_size,
                              hipStream_t stream) {
    (void)in_sizes; (void)n_in; (void)out_size; (void)ws_size;
    const int*   Xe   = (const int*)d_in[0];
    const float* Xc   = (const float*)d_in[1];
    const float* Xl   = (const float*)d_in[2];
    const float* eps  = (const float*)d_in[4];
    const float* emb0 = (const float*)d_in[6];
    const float* emb1 = (const float*)d_in[7];
    auto F = [&](int i) { return (const float*)d_in[i]; };

    // ---- workspace bump allocator (256B aligned)
    char* base = (char*)d_ws; size_t off = 0;
    auto allocB = [&](size_t nbytes) -> void* {
        void* p = base + off; off += (nbytes + 255) & ~(size_t)255; return p;
    };
    auto aU = [&](size_t n) { return (unsigned short*)allocB(n * 2); };
    auto aF = [&](size_t n) { return (float*)allocB(n * 4); };

    // bf16 weights (W stored [N,K] row-major == B^T fragment layout)
    unsigned short* wb_phix1   = aU(512 * 128);
    unsigned short* wb_phix2   = aU(512 * 512);
    unsigned short* wb_phiz    = aU(512 * 512);
    unsigned short* wb_enc1    = aU(512 * 1024);
    unsigned short* wb_enc2    = aU(512 * 512);
    unsigned short* wb_encms   = aU(1024 * 512);   // [mean;std] stacked
    unsigned short* wb_prior   = aU(512 * 512);
    unsigned short* wb_priorms = aU(1024 * 512);
    unsigned short* wb_dec1    = aU(512 * 1024);
    unsigned short* wb_dec2    = aU(512 * 512);
    unsigned short* wb_decms   = aU(256 * 512);
    unsigned short* wb_g0ih    = aU(1536 * 1024);
    unsigned short* wb_g0hh    = aU(1536 * 512);
    unsigned short* wb_g1ih    = aU(1536 * 512);
    unsigned short* wb_g1hh    = aU(1536 * 512);
    float* bf_encms   = aF(1024);
    float* bf_priorms = aF(1024);
    float* bf_decms   = aF(256);
    // activations
    unsigned short* xb     = aU((size_t)SEQL * BATCH * XD);
    unsigned short* tphi   = aU((size_t)SEQL * BATCH * HIDN);
    unsigned short* phix   = aU((size_t)SEQL * BATCH * HIDN);
    unsigned short* t1b    = aU(BATCH * HIDN);
    unsigned short* enc_hb = aU(BATCH * HIDN);
    unsigned short* prior_hb = aU(BATCH * HIDN);
    unsigned short* zb     = aU(BATCH * HIDN);
    unsigned short* phizb  = aU(BATCH * HIDN);
    unsigned short* dec_hb = aU(BATCH * HIDN);
    unsigned short* h0b    = aU(BATCH * HIDN);
    unsigned short* h1b    = aU(BATCH * HIDN);
    float* encms_f   = aF(BATCH * 1024);
    float* priorms_f = aF(BATCH * 1024);
    float* decms_f   = aF(BATCH * 256);
    float* gi_f      = aF(BATCH * 1536);
    float* gh_f      = aF(BATCH * 1536);
    float* h0f       = aF(BATCH * HIDN);
    float* h1f       = aF(BATCH * HIDN);
    float* accv      = aF(8);
    float* dout      = (float*)d_out;

    auto pack = [&](const float* s, unsigned short* d, int n) {
        pack_bf16_k<<<dim3((n + 255) / 256), dim3(256), 0, stream>>>(s, d, n);
    };
    auto cpyf = [&](const float* s, float* d, int n) {
        copy_f32_k<<<dim3((n + 255) / 256), dim3(256), 0, stream>>>(s, d, n);
    };
    auto gemmR = [&](const unsigned short* A0, const unsigned short* A1, int K0, int K,
                     const unsigned short* W, const float* bias,
                     unsigned short* outb, int M, int N) {
        gemm_wmma_bf16<true, true, false><<<dim3(N / 64, M / 64), dim3(256), 0, stream>>>(
            A0, A1, K0, K, W, bias, outb, nullptr, N);
    };
    auto gemmF = [&](const unsigned short* A0, const unsigned short* A1, int K0, int K,
                     const unsigned short* W, const float* bias,
                     float* outf, int M, int N) {
        gemm_wmma_bf16<false, false, true><<<dim3(N / 64, M / 64), dim3(256), 0, stream>>>(
            A0, A1, K0, K, W, bias, nullptr, outf, N);
    };

    // ---- one-time prep (re-run every call; trivially cheap vs GEMM work)
    pack(F(8),  wb_phix1, 512 * 128);
    pack(F(10), wb_phix2, 512 * 512);
    pack(F(12), wb_phiz,  512 * 512);
    pack(F(14), wb_enc1,  512 * 1024);
    pack(F(16), wb_enc2,  512 * 512);
    pack(F(18), wb_encms,             512 * 512);
    pack(F(20), wb_encms + 512 * 512, 512 * 512);
    pack(F(22), wb_prior, 512 * 512);
    pack(F(24), wb_priorms,             512 * 512);
    pack(F(26), wb_priorms + 512 * 512, 512 * 512);
    pack(F(28), wb_dec1,  512 * 1024);
    pack(F(30), wb_dec2,  512 * 512);
    pack(F(32), wb_decms,             128 * 512);
    pack(F(34), wb_decms + 128 * 512, 128 * 512);
    pack(F(36), wb_g0ih, 1536 * 1024);
    pack(F(37), wb_g0hh, 1536 * 512);
    pack(F(38), wb_g1ih, 1536 * 512);
    pack(F(39), wb_g1hh, 1536 * 512);
    cpyf(F(19), bf_encms, 512);        cpyf(F(21), bf_encms + 512, 512);
    cpyf(F(25), bf_priorms, 512);      cpyf(F(27), bf_priorms + 512, 512);
    cpyf(F(33), bf_decms, 128);        cpyf(F(35), bf_decms + 128, 128);
    init_state_k<<<dim3(BATCH * HIDN / 256), dim3(256), 0, stream>>>(h0f, h1f, h0b, h1b, accv);

    // ---- hoisted input path: x and phi_x for ALL timesteps (big parallel GEMMs)
    build_x_k<<<dim3(SEQL * BATCH * XD / 256), dim3(256), 0, stream>>>(Xe, Xc, Xl, emb0, emb1, xb);
    gemmR(xb,   xb,   XD,   XD,   wb_phix1, F(9),  tphi, SEQL * BATCH, HIDN);
    gemmR(tphi, tphi, HIDN, HIDN, wb_phix2, F(11), phix, SEQL * BATCH, HIDN);

    // ---- sequential recurrence
    for (int t = 0; t < SEQL; ++t) {
        const unsigned short* phixt = phix + (size_t)t * BATCH * HIDN;
        // encoder: enc_h = relu(enc2(relu(enc1([phi_x, h1]))))
        gemmR(phixt, h1b, HIDN, 2 * HIDN, wb_enc1, F(15), t1b, BATCH, HIDN);
        gemmR(t1b, t1b, HIDN, HIDN, wb_enc2, F(17), enc_hb, BATCH, HIDN);
        gemmF(enc_hb, enc_hb, HIDN, HIDN, wb_encms, bf_encms, encms_f, BATCH, 2 * HIDN);
        // prior
        gemmR(h1b, h1b, HIDN, HIDN, wb_prior, F(23), prior_hb, BATCH, HIDN);
        gemmF(prior_hb, prior_hb, HIDN, HIDN, wb_priorms, bf_priorms, priorms_f, BATCH, 2 * HIDN);
        // z = eps*softplus(std)+mean ; KLD
        zkld_k<<<dim3(BATCH * HIDN / 256), dim3(256), 0, stream>>>(
            encms_f, priorms_f, eps + (size_t)t * BATCH * HIDN, zb, accv);
        // decoder
        gemmR(zb, zb, HIDN, HIDN, wb_phiz, F(13), phizb, BATCH, HIDN);
        gemmR(phizb, h1b, HIDN, 2 * HIDN, wb_dec1, F(29), t1b, BATCH, HIDN);
        gemmR(t1b, t1b, HIDN, HIDN, wb_dec2, F(31), dec_hb, BATCH, HIDN);
        gemmF(dec_hb, dec_hb, HIDN, HIDN, wb_decms, bf_decms, decms_f, BATCH, 2 * XD);
        nll_k<<<dim3(BATCH * XD / 256), dim3(256), 0, stream>>>(
            decms_f, Xe, Xc, Xl, emb0, emb1, t, accv, dout);
        // GRU layer 0: rnn_in = [phi_x, phi_z]
        gemmF(phixt, phizb, HIDN, 2 * HIDN, wb_g0ih, nullptr, gi_f, BATCH, 3 * HIDN);
        gemmF(h0b, h0b, HIDN, HIDN, wb_g0hh, nullptr, gh_f, BATCH, 3 * HIDN);
        gru_comb_k<<<dim3(BATCH * HIDN / 256), dim3(256), 0, stream>>>(gi_f, gh_f, h0f, h0b);
        // GRU layer 1
        gemmF(h0b, h0b, HIDN, HIDN, wb_g1ih, nullptr, gi_f, BATCH, 3 * HIDN);
        gemmF(h1b, h1b, HIDN, HIDN, wb_g1hh, nullptr, gh_f, BATCH, 3 * HIDN);
        gru_comb_k<<<dim3(BATCH * HIDN / 256), dim3(256), 0, stream>>>(gi_f, gh_f, h1f, h1b);
    }
    fin_k<<<1, 32, 0, stream>>>(accv, dout);
}
